// MaskedMLP_15393162789082
// MI455X (gfx1250) — matrix operand order repaired and checked by reading, not compiled, not run on hardware
//
#include <hip/hip_runtime.h>

typedef __attribute__((ext_vector_type(16))) __bf16 v16bf;
typedef __attribute__((ext_vector_type(8)))  float  v8f;
typedef __attribute__((ext_vector_type(4)))  float  v4f;
typedef __attribute__((vector_size(16)))     int    v4i_vs;

typedef __attribute__((address_space(1))) v4i_vs g_v4i;
typedef __attribute__((address_space(3))) v4i_vs l_v4i;

#define IN_DIM  256
#define OUT_DIM 256
#define KCHUNKS 8    // 256 / 32
#define NTILES  16   // 256 / 16
#define ROWS_PER_BLOCK 128

// Packed B slice per chunk: 16 tiles * 32 lanes * 16 bf16 = 16 KB (hi), 16 KB (lo)
#define CHUNK_BYTES 16384

// workspace layout (bytes):
//   [0, 1024)                 : mask, 256 floats
//   [1024, 1024+131072)       : Bhi packed fragments (256*256 bf16)
//   [1024+131072, +131072)    : Blo packed fragments
#define WS_MASK_OFF 0
#define WS_BHI_OFF  1024
#define WS_BLO_OFF  (1024 + 131072)

#if defined(__AMDGCN__) && __has_builtin(__builtin_amdgcn_global_load_async_to_lds_b128)
#define USE_ASYNC_LDS 1
#else
#define USE_ASYNC_LDS 0
#endif

// ---------------------------------------------------------------------------
// Step 1: mask[o] = (mean(|W[o,:]|) - threshold[o]) > 0 ? 1 : 0
// ---------------------------------------------------------------------------
__global__ void msk_mask_kernel(const float* __restrict__ w,
                                const float* __restrict__ thr,
                                float* __restrict__ mask) {
    int o = blockIdx.x * blockDim.x + threadIdx.x;
    if (o >= OUT_DIM) return;
    float s = 0.0f;
    const float* row = w + (size_t)o * IN_DIM;
    #pragma unroll 8
    for (int k = 0; k < IN_DIM; ++k) s += fabsf(row[k]);
    float v = s * (1.0f / (float)IN_DIM) - thr[o];
    mask[o] = (v > 0.0f) ? 1.0f : 0.0f;
}

// ---------------------------------------------------------------------------
// Step 2: pack masked B = (W*mask)^T into WMMA B-fragment order, bf16 hi/lo.
// B[k][n] = W[n][k] * mask[n].
// Fragment (chunk c, ntile t): lane L holds n = t*16 + L%16, k-half = L/16;
// element i -> k = c*32 + (L/16)*16 + i. Stored contiguously:
//   B*[ ((c*16 + t)*32 + L)*16 + i ]
// ---------------------------------------------------------------------------
__global__ void msk_pack_kernel(const float* __restrict__ w,
                                const float* __restrict__ mask,
                                __bf16* __restrict__ bhi,
                                __bf16* __restrict__ blo) {
    int lane = threadIdx.x;          // 0..31
    int t = blockIdx.x & 15;         // ntile
    int c = blockIdx.x >> 4;         // kchunk
    int n  = t * 16 + (lane & 15);
    int kh = lane >> 4;
    float mv = mask[n];
    size_t base = ((size_t)blockIdx.x * 32 + lane) * 16;
    #pragma unroll
    for (int i = 0; i < 16; ++i) {
        int k = c * 32 + kh * 16 + i;
        float v = w[(size_t)n * IN_DIM + k] * mv;
        __bf16 h = (__bf16)v;
        __bf16 l = (__bf16)(v - (float)h);
        bhi[base + i] = h;
        blo[base + i] = l;
    }
}

// ---------------------------------------------------------------------------
// Step 3: GEMM. Block = 256 threads (8 waves). Each wave: 16 rows x 256 cols.
// B fragments double-buffered in LDS via async global->LDS copies; x streamed
// nontemporally from HBM; split-precision bf16x3 accumulation in f32.
// ---------------------------------------------------------------------------
__global__ __launch_bounds__(256) void msk_gemm_kernel(
        const float* __restrict__ x,
        const char* __restrict__ bhi,
        const char* __restrict__ blo,
        float* __restrict__ out) {
    // smem[buf][seg][16KB] : seg 0 = Bhi chunk, seg 1 = Blo chunk
    __shared__ alignas(1024) char smem[2][2][CHUNK_BYTES];

    const int tid  = threadIdx.x;
    const int lane = tid & 31;
    const int wid  = tid >> 5;
    const int lh   = lane >> 4;      // lane half: 0 or 1
    const int ln   = lane & 15;      // 0..15

    const int row0 = blockIdx.x * ROWS_PER_BLOCK + wid * 16;
    const float* xrow = x + (size_t)(row0 + ln) * IN_DIM;

    // -------- staging: each thread copies 128 B of the 32 KB chunk slice ----
    const int seg  = tid >> 7;                 // 0: hi, 1: lo (128 threads each)
    const int soff = (tid & 127) * 128;
    const char* gseg = seg ? blo : bhi;
    char* lseg = &smem[0][seg][soff];          // buffer 0; buffer 1 = +32 KB

    auto stage = [&](int c, int buf) {
        const char* gp = gseg + (size_t)c * CHUNK_BYTES + soff;
        char* lp = lseg + buf * (2 * CHUNK_BYTES);
#if USE_ASYNC_LDS
        #pragma unroll
        for (int i = 0; i < 8; ++i) {
            __builtin_amdgcn_global_load_async_to_lds_b128(
                (g_v4i*)(gp + 16 * i),
                (l_v4i*)(lp + 16 * i),
                0, 0);
        }
#else
        #pragma unroll
        for (int i = 0; i < 8; ++i) {
            v4f tmp = *(const v4f*)(gp + 16 * i);
            *(v4f*)(lp + 16 * i) = tmp;
        }
#endif
    };

    v8f acc[NTILES];
    #pragma unroll
    for (int t = 0; t < NTILES; ++t) acc[t] = (v8f){};

    stage(0, 0);

    for (int c = 0; c < KCHUNKS; ++c) {
        const int buf = c & 1;
#if USE_ASYNC_LDS
#if __has_builtin(__builtin_amdgcn_s_wait_asynccnt)
        __builtin_amdgcn_s_wait_asynccnt(0);
#else
        asm volatile("s_wait_asynccnt 0" ::: "memory");
#endif
#endif
        __syncthreads();
        if (c + 1 < KCHUNKS) stage(c + 1, buf ^ 1);

        // ---- load A tile (16 rows x 32 k) in contiguous k order (matches packed B) ----
        // element i -> k = c*32 + lh*16 + i
        float f[16];
        {
            const v4f* p0 = (const v4f*)(xrow + c * 32 + lh * 16);       // g=0
            const v4f* p1 = (const v4f*)(xrow + c * 32 + lh * 16 + 8);  // g=1
            v4f a0 = __builtin_nontemporal_load(p0);
            v4f a1 = __builtin_nontemporal_load(p0 + 1);
            v4f b0 = __builtin_nontemporal_load(p1);
            v4f b1 = __builtin_nontemporal_load(p1 + 1);
            #pragma unroll
            for (int j = 0; j < 4; ++j) {
                f[j]      = a0[j];
                f[4 + j]  = a1[j];
                f[8 + j]  = b0[j];
                f[12 + j] = b1[j];
            }
        }
        v16bf ah, al;
        #pragma unroll
        for (int i = 0; i < 16; ++i) {
            __bf16 h = (__bf16)f[i];
            ah[i] = h;
            al[i] = (__bf16)(f[i] - (float)h);
        }

        // ---- 16 n-tiles, 3 WMMAs each, B fragments from LDS ----
        const v16bf* sh = (const v16bf*)&smem[buf][0][lane * 32];
        const v16bf* sl = (const v16bf*)&smem[buf][1][lane * 32];
        #pragma unroll
        for (int t = 0; t < NTILES; ++t) {
            v16bf bh = sh[t * 32];     // t*1024 bytes
            v16bf bl = sl[t * 32];
            acc[t] = __builtin_amdgcn_wmma_f32_16x16x32_bf16(
                false, ah, false, bh, (short)0, acc[t], false, false);
            acc[t] = __builtin_amdgcn_wmma_f32_16x16x32_bf16(
                false, al, false, bh, (short)0, acc[t], false, false);
            acc[t] = __builtin_amdgcn_wmma_f32_16x16x32_bf16(
                false, ah, false, bl, (short)0, acc[t], false, false);
        }
        __syncthreads();
    }

    // ---- store: C layout -> VGPR v holds row m = v + 8*lh, col n = ln ----
    #pragma unroll
    for (int t = 0; t < NTILES; ++t) {
        #pragma unroll
        for (int v = 0; v < 8; ++v) {
            int r = row0 + lh * 8 + v;
            __builtin_nontemporal_store(acc[t][v],
                                        &out[(size_t)r * OUT_DIM + t * 16 + ln]);
        }
    }
}

// ---------------------------------------------------------------------------
extern "C" void kernel_launch(void* const* d_in, const int* in_sizes, int n_in,
                              void* d_out, int out_size, void* d_ws, size_t ws_size,
                              hipStream_t stream) {
    const float* x   = (const float*)d_in[0];
    const float* w   = (const float*)d_in[1];
    const float* thr = (const float*)d_in[2];
    float* out = (float*)d_out;

    float*  mask = (float*)((char*)d_ws + WS_MASK_OFF);
    __bf16* bhi  = (__bf16*)((char*)d_ws + WS_BHI_OFF);
    __bf16* blo  = (__bf16*)((char*)d_ws + WS_BLO_OFF);

    const int B = in_sizes[0] / IN_DIM;  // 262144

    msk_mask_kernel<<<1, 256, 0, stream>>>(w, thr, mask);
    msk_pack_kernel<<<KCHUNKS * NTILES, 32, 0, stream>>>(w, mask, bhi, blo);
    msk_gemm_kernel<<<B / ROWS_PER_BLOCK, 256, 0, stream>>>(
        x, (const char*)bhi, (const char*)blo, out);
}
